// GIN_82308753261080
// MI455X (gfx1250) — compile-verified, matrix operand order checked by reading, not run on hardware
//
#include <hip/hip_runtime.h>
#include <hip/hip_bf16.h>

#define HDIM 128
#define NNODES 100000
#define NEDGES 1600000
#define NGRAPHS 512
#define BN_EPS 1e-5f

typedef __attribute__((ext_vector_type(16))) __bf16 v16bf;
typedef __attribute__((ext_vector_type(8)))  float  v8f;

// ---------- helpers ----------
__device__ __forceinline__ unsigned short f2bf(float f) {
    union { float f; unsigned int u; } v; v.f = f;
    unsigned int u = v.u;
    unsigned int r = (u + 0x7FFFu + ((u >> 16) & 1u)) >> 16;   // RNE
    return (unsigned short)r;
}

// ---------- layer 1 (4 -> 128) ----------
__global__ void scatter_add4(const int* __restrict__ src, const int* __restrict__ dst,
                             const float* __restrict__ x, float* __restrict__ agg4) {
    int e = blockIdx.x * blockDim.x + threadIdx.x;
    if (e >= NEDGES) return;
    int s = src[e], d = dst[e];
    float4 v = reinterpret_cast<const float4*>(x)[s];
    float* ap = agg4 + (size_t)d * 4;
    atomicAdd(ap + 0, v.x); atomicAdd(ap + 1, v.y);
    atomicAdd(ap + 2, v.z); atomicAdd(ap + 3, v.w);
}

__global__ void lin1(const float* __restrict__ x, const float* __restrict__ agg4,
                     const float* __restrict__ W1, const float* __restrict__ b1,
                     float* __restrict__ z) {
    size_t i = (size_t)blockIdx.x * blockDim.x + threadIdx.x;
    size_t node = i >> 7;
    int ch = (int)(i & (HDIM - 1));
    if (node >= NNODES) return;
    float4 xv = reinterpret_cast<const float4*>(x)[node];
    float4 av = reinterpret_cast<const float4*>(agg4)[node];
    float s0 = xv.x + av.x, s1 = xv.y + av.y, s2 = xv.z + av.z, s3 = xv.w + av.w;
    float acc = b1[ch];
    acc = fmaf(s0, W1[0 * HDIM + ch], acc);
    acc = fmaf(s1, W1[1 * HDIM + ch], acc);
    acc = fmaf(s2, W1[2 * HDIM + ch], acc);
    acc = fmaf(s3, W1[3 * HDIM + ch], acc);
    z[i] = acc;
}

// ---------- H=128 scatter: one wave per edge, float4 per lane ----------
__global__ void scatter_add128(const int* __restrict__ src, const int* __restrict__ dst,
                               const float* __restrict__ h, float* __restrict__ agg) {
    size_t gid = (size_t)blockIdx.x * blockDim.x + threadIdx.x;
    size_t edge = gid >> 5;
    int lane = (int)(gid & 31);
    if (edge >= NEDGES) return;
    int s = src[edge], d = dst[edge];
    float4 v = reinterpret_cast<const float4*>(h + (size_t)s * HDIM)[lane];
    float* ap = agg + (size_t)d * HDIM + lane * 4;
    atomicAdd(ap + 0, v.x); atomicAdd(ap + 1, v.y);
    atomicAdd(ap + 2, v.z); atomicAdd(ap + 3, v.w);
}

// ---------- s = h + agg, cast to bf16 staging ----------
__global__ void add_to_bf16(const float* __restrict__ h, const float* __restrict__ agg,
                            unsigned short* __restrict__ out, int n) {
    int i = blockIdx.x * blockDim.x + threadIdx.x;
    if (i < n) out[i] = f2bf(h[i] + agg[i]);
}

// ---------- weight prep: W[k][n] f32 -> Wt[n][k] bf16 ----------
__global__ void prep_w(const float* __restrict__ W, unsigned short* __restrict__ Wt) {
    int i = blockIdx.x * blockDim.x + threadIdx.x;      // 128*128
    int k = i >> 7, n = i & (HDIM - 1);
    Wt[n * HDIM + k] = f2bf(W[k * HDIM + n]);
}

// ---------- WMMA GEMM: Z[100000x128] = A(bf16) @ W(bf16,transposed) + bias ----------
// grid = 6250 blocks x 256 threads; wave w of a block owns C tile (blockIdx*16, w*16)
__global__ void gemm_wmma(const unsigned short* __restrict__ Abf,
                          const unsigned short* __restrict__ Wt,
                          const float* __restrict__ bias,
                          float* __restrict__ Z) {
    int wave = threadIdx.x >> 5;            // 0..7 -> N tile
    int lane = threadIdx.x & 31;
    int tileM = blockIdx.x * 16;
    int tileN = wave * 16;
    int r16 = lane & 15;
    int khalf = (lane >> 4) * 16;

    const __bf16* arow = reinterpret_cast<const __bf16*>(Abf) + (size_t)(tileM + r16) * HDIM + khalf;
    const __bf16* brow = reinterpret_cast<const __bf16*>(Wt)  + (size_t)(tileN + r16) * HDIM + khalf;

    v8f acc = {};
#pragma unroll
    for (int kk = 0; kk < HDIM; kk += 32) {
        v16bf a = *reinterpret_cast<const v16bf*>(arow + kk);
        v16bf b = *reinterpret_cast<const v16bf*>(brow + kk);
        acc = __builtin_amdgcn_wmma_f32_16x16x32_bf16(false, a, false, b,
                                                      (short)0, acc, false, false);
    }
    int col = tileN + r16;
    float bv = bias[col];
    int rbase = tileM + (lane >> 4) * 8;    // C layout: VGPR r -> row r (lanes 0-15) / r+8 (lanes 16-31)
#pragma unroll
    for (int r = 0; r < 8; ++r)
        Z[(size_t)(rbase + r) * HDIM + col] = acc[r] + bv;
}

// ---------- BatchNorm: per-channel sum / sumsq ----------
__global__ void bn_stats(const float* __restrict__ z, float* __restrict__ stats) {
    int ch = threadIdx.x;                   // blockDim = 128
    float s = 0.f, s2 = 0.f;
    for (int node = blockIdx.x; node < NNODES; node += gridDim.x) {
        float v = z[(size_t)node * HDIM + ch];
        s += v; s2 += v * v;
    }
    atomicAdd(&stats[ch], s);
    atomicAdd(&stats[HDIM + ch], s2);
}

__global__ void bn_finalize(const float* __restrict__ stats,
                            const float* __restrict__ g, const float* __restrict__ be,
                            float* __restrict__ ss) {
    int ch = threadIdx.x;                   // one block of 128
    float inv_n = 1.0f / (float)NNODES;
    float mu = stats[ch] * inv_n;
    float var = stats[HDIM + ch] * inv_n - mu * mu;
    float sc = g[ch] * rsqrtf(var + BN_EPS);
    ss[ch] = sc;
    ss[HDIM + ch] = be[ch] - mu * sc;
}

__global__ void bn_apply_relu(const float* __restrict__ z, const float* __restrict__ ss,
                              float* __restrict__ hout, int n) {
    int i = blockIdx.x * blockDim.x + threadIdx.x;
    if (i >= n) return;
    int ch = i & (HDIM - 1);
    float v = fmaf(z[i], ss[ch], ss[HDIM + ch]);
    hout[i] = v > 0.f ? v : 0.f;
}

// ---------- pooling + head ----------
__global__ void pool_sum(const float* __restrict__ h, const int* __restrict__ batch,
                         float* __restrict__ pooled, float* __restrict__ counts) {
    size_t i = (size_t)blockIdx.x * blockDim.x + threadIdx.x;
    size_t node = i >> 7;
    int ch = (int)(i & (HDIM - 1));
    if (node >= NNODES) return;
    int g = batch[node];
    atomicAdd(&pooled[(size_t)g * HDIM + ch], h[node * HDIM + ch]);
    if (ch == 0) atomicAdd(&counts[g], 1.0f);
}

__global__ void final_head(const float* __restrict__ pooled, const float* __restrict__ counts,
                           const float* __restrict__ Wf, const float* __restrict__ bf_,
                           float* __restrict__ out) {
    int g = blockIdx.x * (blockDim.x >> 5) + (threadIdx.x >> 5);
    int lane = threadIdx.x & 31;
    if (g >= NGRAPHS) return;
    float acc = 0.f;
#pragma unroll
    for (int c = lane; c < HDIM; c += 32)
        acc += pooled[(size_t)g * HDIM + c] * Wf[c];
#pragma unroll
    for (int off = 16; off > 0; off >>= 1)
        acc += __shfl_down(acc, off, 32);
    if (lane == 0) {
        float cnt = fmaxf(counts[g], 1.0f);
        float v = acc / cnt + bf_[0];
        out[g] = 1.0f / (1.0f + __expf(-v));
    }
}

// ---------- host driver ----------
static inline size_t align256(size_t x) { return (x + 255) & ~(size_t)255; }

extern "C" void kernel_launch(void* const* d_in, const int* in_sizes, int n_in,
                              void* d_out, int out_size, void* d_ws, size_t ws_size,
                              hipStream_t stream) {
    const float* x    = (const float*)d_in[0];
    const int*   ei   = (const int*)d_in[1];         // [2][NEDGES] (int32 under JAX x32)
    const int*   bat  = (const int*)d_in[2];
    const float* W1 = (const float*)d_in[3],  *b1 = (const float*)d_in[4];
    const float* g1 = (const float*)d_in[5],  *be1 = (const float*)d_in[6];
    const float* W2 = (const float*)d_in[7],  *b2 = (const float*)d_in[8];
    const float* g2 = (const float*)d_in[9],  *be2 = (const float*)d_in[10];
    const float* W3 = (const float*)d_in[11], *b3 = (const float*)d_in[12];
    const float* g3 = (const float*)d_in[13], *be3 = (const float*)d_in[14];
    const float* Wf = (const float*)d_in[15], *bf_ = (const float*)d_in[16];
    float* out = (float*)d_out;

    const int* src = ei;
    const int* dst = ei + NEDGES;

    // workspace carve-up
    char* p = (char*)d_ws;
    const size_t featBytes = (size_t)NNODES * HDIM * sizeof(float);   // 51.2 MB
    float* hA   = (float*)p;                 p += align256(featBytes);
    float* hB   = (float*)p;                 p += align256(featBytes);
    float* agg  = (float*)p;                 p += align256(featBytes); // doubles as z
    unsigned short* abf = (unsigned short*)p; p += align256((size_t)NNODES * HDIM * 2);
    float* agg4 = (float*)p;                 p += align256((size_t)NNODES * 4 * sizeof(float));
    unsigned short* wt2 = (unsigned short*)p; p += align256(HDIM * HDIM * 2);
    unsigned short* wt3 = (unsigned short*)p; p += align256(HDIM * HDIM * 2);
    float* stats = (float*)p;                p += align256(2 * HDIM * sizeof(float));
    float* ss    = (float*)p;                p += align256(2 * HDIM * sizeof(float));
    float* pooled = (float*)p;               p += align256((size_t)NGRAPHS * HDIM * sizeof(float));
    float* counts = (float*)p;               p += align256(NGRAPHS * sizeof(float));

    const int TPB = 256;
    const int nFeat = NNODES * HDIM;                        // 12.8M
    const int featBlocks = (nFeat + TPB - 1) / TPB;         // 50000
    const int edgeBlocks = (NEDGES + TPB - 1) / TPB;        // 6250
    const int edgeWaveBlocks = (int)(((size_t)NEDGES * 32 + TPB - 1) / TPB); // 200000
    const int gemmBlocks = NNODES / 16;                     // 6250 (exact)

    // weight prep (independent)
    prep_w<<<(HDIM * HDIM) / TPB, TPB, 0, stream>>>(W2, wt2);
    prep_w<<<(HDIM * HDIM) / TPB, TPB, 0, stream>>>(W3, wt3);

    // ---- layer 1: 4 -> 128 ----
    hipMemsetAsync(agg4, 0, (size_t)NNODES * 4 * sizeof(float), stream);
    hipMemsetAsync(stats, 0, 2 * HDIM * sizeof(float), stream);
    scatter_add4<<<edgeBlocks, TPB, 0, stream>>>(src, dst, x, agg4);
    lin1<<<featBlocks, TPB, 0, stream>>>(x, agg4, W1, b1, agg);          // z -> agg
    bn_stats<<<512, HDIM, 0, stream>>>(agg, stats);
    bn_finalize<<<1, HDIM, 0, stream>>>(stats, g1, be1, ss);
    bn_apply_relu<<<featBlocks, TPB, 0, stream>>>(agg, ss, hA, nFeat);

    // ---- layer 2 ----
    hipMemsetAsync(agg, 0, featBytes, stream);
    hipMemsetAsync(stats, 0, 2 * HDIM * sizeof(float), stream);
    scatter_add128<<<edgeWaveBlocks, TPB, 0, stream>>>(src, dst, hA, agg);
    add_to_bf16<<<featBlocks, TPB, 0, stream>>>(hA, agg, abf, nFeat);
    gemm_wmma<<<gemmBlocks, TPB, 0, stream>>>(abf, wt2, b2, agg);        // z -> agg
    bn_stats<<<512, HDIM, 0, stream>>>(agg, stats);
    bn_finalize<<<1, HDIM, 0, stream>>>(stats, g2, be2, ss);
    bn_apply_relu<<<featBlocks, TPB, 0, stream>>>(agg, ss, hB, nFeat);

    // ---- layer 3 ----
    hipMemsetAsync(agg, 0, featBytes, stream);
    hipMemsetAsync(stats, 0, 2 * HDIM * sizeof(float), stream);
    scatter_add128<<<edgeWaveBlocks, TPB, 0, stream>>>(src, dst, hB, agg);
    add_to_bf16<<<featBlocks, TPB, 0, stream>>>(hB, agg, abf, nFeat);
    gemm_wmma<<<gemmBlocks, TPB, 0, stream>>>(abf, wt3, b3, agg);        // z -> agg
    bn_stats<<<512, HDIM, 0, stream>>>(agg, stats);
    bn_finalize<<<1, HDIM, 0, stream>>>(stats, g3, be3, ss);
    bn_apply_relu<<<featBlocks, TPB, 0, stream>>>(agg, ss, hA, nFeat);

    // ---- pooling + head ----
    hipMemsetAsync(pooled, 0, (size_t)NGRAPHS * HDIM * sizeof(float), stream);
    hipMemsetAsync(counts, 0, NGRAPHS * sizeof(float), stream);
    pool_sum<<<featBlocks, TPB, 0, stream>>>(hA, bat, pooled, counts);
    final_head<<<NGRAPHS / 8, TPB, 0, stream>>>(pooled, counts, Wf, bf_, out);
    (void)n_in; (void)in_sizes; (void)out_size; (void)ws_size;
}
